// GNN_Network_spheroid_14181982011844
// MI455X (gfx1250) — matrix operand order repaired and checked by reading, not compiled, not run on hardware
//
#include <hip/hip_runtime.h>
#include <math.h>

// ---------------------------------------------------------------------------
// GNN spheroid network for MI455X (gfx1250, wave32).
// Edge-MLP (hot loop: 10 layers x 320k edges x [68..78]->15) runs on
// v_wmma_f32_16x16x32_f16: 16 edges per wave-private LDS tile, K padded to 96
// (3 WMMA steps). Weight tile is pre-transposed/converted to f16 once per
// layer so each lane's B fragment is one contiguous 32B register load, kept
// loop-invariant across a persistent wave-strided tile loop.
// ---------------------------------------------------------------------------

#define KQ      10   // top-K neighbors
#define NHEADS  5
#define NLAT    15
#define NHCH    20   // node hidden (NH)

typedef __attribute__((ext_vector_type(16))) _Float16 v16h;
typedef __attribute__((ext_vector_type(8)))  _Float16 v8h;
typedef __attribute__((ext_vector_type(8)))  float    v8f;

// ------------------------------ embed MLP ----------------------------------
__global__ void embed_kernel(const float* __restrict__ x, int n, int din,
                             const float* __restrict__ w1, const float* __restrict__ b1,
                             const float* __restrict__ a1p,
                             const float* __restrict__ w2, const float* __restrict__ b2,
                             const float* __restrict__ a2p,
                             float* __restrict__ out) {
  int i = blockIdx.x * blockDim.x + threadIdx.x;
  if (i >= n) return;
  const float a1 = a1p[0], a2 = a2p[0];
  float xi[9];
  for (int j = 0; j < din; ++j) xi[j] = x[i * din + j];
  float hmid[30];
  for (int o = 0; o < 30; ++o) {
    float acc = b1[o];
    for (int j = 0; j < din; ++j) acc += xi[j] * w1[j * 30 + o];
    hmid[o] = acc >= 0.f ? acc : a1 * acc;
  }
  for (int o = 0; o < 30; ++o) {
    float acc = b2[o];
    for (int j = 0; j < 30; ++j) acc += hmid[j] * w2[j * 30 + o];
    out[i * 30 + o] = acc >= 0.f ? acc : a2 * acc;
  }
}

// ------------------------------ degree count -------------------------------
__global__ void count_kernel(const int* __restrict__ dst, int E, float* __restrict__ cnt) {
  int e = blockIdx.x * blockDim.x + threadIdx.x;
  if (e < E) atomicAdd(&cnt[dst[e]], 1.0f);
}

// ------------------------------ global pool --------------------------------
__global__ void pool_kernel(const float* __restrict__ h, int n, int din,
                            const float* __restrict__ wg, const float* __restrict__ bg,
                            const float* __restrict__ a3p, float* __restrict__ gp_sum) {
  __shared__ float red[3];
  if (threadIdx.x < 3) red[threadIdx.x] = 0.f;
  __syncthreads();
  int i = blockIdx.x * blockDim.x + threadIdx.x;
  if (i < n) {
    const float a3 = a3p[0];
    for (int c = 0; c < 3; ++c) {
      float acc = bg[c];
      for (int j = 0; j < din; ++j) acc += h[i * din + j] * wg[j * 3 + c];
      acc = acc >= 0.f ? acc : a3 * acc;
      atomicAdd(&red[c], acc);
    }
  }
  __syncthreads();
  if (threadIdx.x < 3) atomicAdd(&gp_sum[threadIdx.x], red[threadIdx.x]);
}

// --------------------- weight prep: w1 -> padded f16, N-major --------------
// Bt[n][k] = w1[k*15+n] for k<KTOT, n<15; zero-padded to 16 x 96.
__global__ void prep_w1_kernel(const float* __restrict__ w1, int KTOT,
                               _Float16* __restrict__ Bt) {
  int i = blockIdx.x * blockDim.x + threadIdx.x;
  if (i >= 16 * 96) return;
  int n = i / 96, k = i % 96;
  float v = (k < KTOT && n < 15) ? w1[k * 15 + n] : 0.0f;
  Bt[n * 96 + k] = (_Float16)v;
}

// ------------------------------ edge MLP (WMMA) ----------------------------
// Per edge e (src -> dst):
//   ef   = prelu((pos[dst]-pos[src]) @ we + be, a4)                  [15]
//   feat = [h[src](DIN) | m[src](30) | gp(3) | ef(15)]               [DIN+48]
//   msg  = prelu(feat @ w1 + b1, a1);   scatter-add msg into sacc[dst].
// Wave-private 16x96 f16 A tiles in LDS; B fragments register-resident.
template <int DIN>
__global__ __launch_bounds__(256) void edge_msg_kernel(
    const float* __restrict__ h,
    const float* __restrict__ m,
    const float* __restrict__ pos,
    const int* __restrict__ edges, int E, int ntiles,
    const float* __restrict__ gp_sum, float invN,
    const float* __restrict__ we, const float* __restrict__ be,
    const float* __restrict__ a4p,
    const _Float16* __restrict__ Bt,   // [16][96] transposed padded f16 w1
    const float* __restrict__ b1, const float* __restrict__ a1p,
    float* __restrict__ sacc) {
  constexpr int KTOT = DIN + 48;
  __shared__ __align__(16) _Float16 As[8][16][96];   // per-wave edge tiles

  const int tid     = threadIdx.x;
  const int wv      = tid >> 5;        // wave in block (0..7)
  const int lane    = tid & 31;
  const int el      = lane >> 1;       // edge-in-tile (0..15) for staging
  const int part    = lane & 1;        // which 48-wide K half this lane stages
  const int halfsel = lane >> 4;       // WMMA K-half select
  const int rc      = lane & 15;       // A row (M) == C/D column (N)

  const float a4 = a4p[0], a1 = a1p[0];
  const float g0 = gp_sum[0] * invN, g1 = gp_sum[1] * invN, g2 = gp_sum[2] * invN;
  const float bb = b1[rc < 15 ? rc : 0];

  // B fragments: lane needs K = 32*kc + 16*halfsel + i, N = rc. In N-major
  // f16 storage that is 32 contiguous bytes -> one v16h load per chunk,
  // loop-invariant for the whole kernel.
  v16h bfrag[3];
#pragma unroll
  for (int kc = 0; kc < 3; ++kc)
    bfrag[kc] = *reinterpret_cast<const v16h*>(Bt + rc * 96 + kc * 32 + halfsel * 16);

  const int gwave  = blockIdx.x * 8 + wv;
  const int nwaves = gridDim.x * 8;

  for (int tile = gwave; tile < ntiles; tile += nwaves) {
    const int ebase = tile * 16;
    const int edge  = ebase + el;

    // ---- stage this wave's 16 x 96 A tile (f16, pair-packed stores) ----
    if (edge < E) {
      const int src  = edges[edge];
      const int dstn = edges[E + edge];
      const float dx = pos[dstn * 3 + 0] - pos[src * 3 + 0];
      const float dy = pos[dstn * 3 + 1] - pos[src * 3 + 1];
      const float dz = pos[dstn * 3 + 2] - pos[src * 3 + 2];
      auto featval = [&](int j) -> float {
        if (j < DIN)            return h[src * DIN + j];
        if (j < DIN + 30)       return m[src * 30 + (j - DIN)];
        if (j == DIN + 30)      return g0;
        if (j == DIN + 31)      return g1;
        if (j == DIN + 32)      return g2;
        if (j < KTOT) {
          int c = j - (DIN + 33);
          float t = dx * we[c] + dy * we[15 + c] + dz * we[30 + c] + be[c];
          return t >= 0.f ? t : a4 * t;
        }
        return 0.f;
      };
      auto stage_pair = [&](int j) {
        union { _Float16 f[2]; unsigned int u; } pk;
        pk.f[0] = (_Float16)featval(j);
        pk.f[1] = (_Float16)featval(j + 1);
        *reinterpret_cast<unsigned int*>(&As[wv][el][j]) = pk.u;
      };
      if (part == 0) {
#pragma unroll
        for (int jj = 0; jj < 48; jj += 2) stage_pair(jj);
      } else {
#pragma unroll
        for (int jj = 0; jj < 48; jj += 2) stage_pair(48 + jj);
      }
    } else {
      const int j0 = part * 48;
#pragma unroll
      for (int jj = 0; jj < 48; jj += 2)
        *reinterpret_cast<unsigned int*>(&As[wv][el][j0 + jj]) = 0u;
    }

    // Tile is wave-private: DS ops from one wave execute in order, so no
    // block barrier is needed; fence the compiler + drain LDS explicitly.
    asm volatile("s_wait_dscnt 0x0" ::: "memory");

    // ---- 3 x wmma_f32_16x16x32_f16 over padded K = 96 ----
    v8f acc = {};
#pragma unroll
    for (int kc = 0; kc < 3; ++kc) {
      union { v16h v; v8h half8[2]; } au;
      const _Float16* arow = &As[wv][rc][0];
      au.half8[0] = *reinterpret_cast<const v8h*>(arow + kc * 32 + halfsel * 16);
      au.half8[1] = *reinterpret_cast<const v8h*>(arow + kc * 32 + halfsel * 16 + 8);
      acc = __builtin_amdgcn_wmma_f32_16x16x32_f16(
          false, au.v, false, bfrag[kc], (short)0, acc, false, false);
    }

    // ---- bias + PReLU + atomic scatter-add into sacc[dst] ----
    if (rc < 15) {
#pragma unroll
      for (int r = 0; r < 8; ++r) {
        int mrow = r + 8 * halfsel;     // C/D row M for VGPR r
        int e2 = ebase + mrow;
        if (e2 < E) {
          int dstn = edges[E + e2];
          float v = acc[r] + bb;
          v = v >= 0.f ? v : a1 * v;
          atomicAdd(&sacc[dstn * 15 + rc], v);
        }
      }
    }
  }
}

// ------------------------------ node update --------------------------------
__global__ void node_update_kernel(const float* __restrict__ h,
                                   const float* __restrict__ m,
                                   const float* __restrict__ s,
                                   const float* __restrict__ cnt,
                                   const float* __restrict__ gp_sum, float invN,
                                   int n, int din,
                                   const float* __restrict__ w2,
                                   const float* __restrict__ b2,
                                   const float* __restrict__ a2p,
                                   const float* __restrict__ resid,
                                   float* __restrict__ out) {
  int i = blockIdx.x * blockDim.x + threadIdx.x;
  if (i >= n) return;
  const float a2 = a2p[0];
  float xb[78];
  for (int j = 0; j < din; ++j) xb[j] = h[i * din + j];
  for (int j = 0; j < 30; ++j)  xb[din + j] = m[i * 30 + j];
  float c = cnt[i]; c = c > 1.0f ? c : 1.0f;
  float cinv = 1.0f / c;
  for (int j = 0; j < 15; ++j)  xb[din + 30 + j] = s[i * 15 + j] * cinv;
  for (int j = 0; j < 3; ++j)   xb[din + 45 + j] = gp_sum[j] * invN;
  const int KT = din + 48;
  for (int o = 0; o < NHCH; ++o) {
    float acc = b2[o];
    for (int j = 0; j < KT; ++j) acc += xb[j] * w2[j * NHCH + o];
    acc = acc >= 0.f ? acc : a2 * acc;
    if (resid) acc += resid[i * NHCH + o];
    out[i * NHCH + o] = acc;
  }
}

// ------------------------------ brute-force kNN ----------------------------
__global__ __launch_bounds__(32) void knn_kernel(const float* __restrict__ xq,
                                                 const float* __restrict__ xc,
                                                 int nq, int n,
                                                 int* __restrict__ knnIdx) {
  int q = blockIdx.x;
  if (q >= nq) return;
  int lane = threadIdx.x;
  float qx = xq[q * 3], qy = xq[q * 3 + 1], qz = xq[q * 3 + 2];
  float bd[KQ]; int bi[KQ];
  for (int k = 0; k < KQ; ++k) { bd[k] = 3.4e38f; bi[k] = 0; }
  for (int i = lane; i < n; i += 32) {
    float dx = qx - xc[i * 3], dy = qy - xc[i * 3 + 1], dz = qz - xc[i * 3 + 2];
    float d = dx * dx + dy * dy + dz * dz;
    int wk = 0; float wvv = bd[0];
    for (int k = 1; k < KQ; ++k) if (bd[k] > wvv) { wvv = bd[k]; wk = k; }
    if (d < wvv) { bd[wk] = d; bi[wk] = i; }
  }
  __shared__ float sd[32 * KQ];
  __shared__ int   si[32 * KQ];
  for (int k = 0; k < KQ; ++k) { sd[lane * KQ + k] = bd[k]; si[lane * KQ + k] = bi[k]; }
  __syncthreads();
  if (lane == 0) {
    for (int k = 0; k < KQ; ++k) {
      int best = 0; float bvv = sd[0];
      for (int t = 1; t < 32 * KQ; ++t) if (sd[t] < bvv) { bvv = sd[t]; best = t; }
      knnIdx[q * KQ + k] = si[best];
      sd[best] = 3.4e38f;
    }
  }
}

// ------------------------------ attention decode ---------------------------
__global__ __launch_bounds__(128) void attention_kernel(
    const float* __restrict__ hfeat,   // [N, 20]
    const float* __restrict__ xq,      // [NQ, 3]
    const float* __restrict__ pos,     // [N, 3]
    const int* __restrict__ knnIdx,    // [NQ, 10]
    const float* __restrict__ pv, const float* __restrict__ we,
    const float* __restrict__ be, const float* __restrict__ wc,
    const float* __restrict__ bc, const float* __restrict__ wvp,
    const float* __restrict__ bvv, const float* __restrict__ w1,
    const float* __restrict__ b1, const float* __restrict__ w2,
    const float* __restrict__ b2,
    const float* __restrict__ a1p, const float* __restrict__ a2p,
    const float* __restrict__ a3p,
    int nq, float* __restrict__ out) {
  int q = blockIdx.x;
  if (q >= nq) return;
  int t = threadIdx.x;
  __shared__ int   sidx[KQ];
  __shared__ float shc[KQ][35];
  __shared__ float sval[KQ][NHEADS * NLAT];
  __shared__ float spre[KQ][NHEADS];
  __shared__ float salpha[KQ][NHEADS];
  __shared__ float so[NLAT];
  __shared__ float sy[NLAT];
  const float a1 = a1p[0], a2 = a2p[0], a3 = a3p[0];
  if (t < KQ) sidx[t] = knnIdx[q * KQ + t];
  __syncthreads();
  const float qx = xq[q * 3], qy = xq[q * 3 + 1], qz = xq[q * 3 + 2];
  for (int idx = t; idx < KQ * 35; idx += 128) {
    int k = idx / 35, j = idx % 35;
    int node = sidx[k];
    float v;
    if (j < NHCH) {
      v = hfeat[node * NHCH + j];
    } else {
      int c = j - NHCH;
      float dx = qx - pos[node * 3], dy = qy - pos[node * 3 + 1], dz = qz - pos[node * 3 + 2];
      float tt = dx * we[c] + dy * we[15 + c] + dz * we[30 + c] + be[c];
      v = tt >= 0.f ? tt : a3 * tt;
    }
    shc[k][j] = v;
  }
  __syncthreads();
  if (t < KQ * NHEADS) {
    int k = t / NHEADS, hd = t % NHEADS;
    float sc = 0.f;
    for (int l = 0; l < NLAT; ++l) {
      int col = hd * NLAT + l;
      float cx = bc[col], vx = bvv[col];
      for (int j = 0; j < 35; ++j) {
        float hv = shc[k][j];
        cx += hv * wc[j * (NHEADS * NLAT) + col];
        vx += hv * wvp[j * (NHEADS * NLAT) + col];
      }
      sval[k][col] = vx;
      sc += pv[col] * cx;
    }
    sc *= 0.2581988897471611f;  // 1/sqrt(NLAT)
    spre[k][hd] = sc >= 0.f ? sc : a1 * sc;
  }
  __syncthreads();
  if (t < NHEADS) {
    float mx = -3.4e38f;
    for (int k = 0; k < KQ; ++k) mx = fmaxf(mx, spre[k][t]);
    float sm = 0.f;
    for (int k = 0; k < KQ; ++k) { float e = __expf(spre[k][t] - mx); salpha[k][t] = e; sm += e; }
    float inv = 1.f / sm;
    for (int k = 0; k < KQ; ++k) salpha[k][t] *= inv;
  }
  __syncthreads();
  if (t < NLAT) {
    float o = 0.f;
    for (int hd = 0; hd < NHEADS; ++hd)
      for (int k = 0; k < KQ; ++k)
        o += salpha[k][hd] * sval[k][hd * NLAT + t];
    so[t] = o * (1.0f / NHEADS);
  }
  __syncthreads();
  if (t < NLAT) {
    float acc = b1[t];
    for (int l = 0; l < NLAT; ++l) acc += so[l] * w1[l * NLAT + t];
    sy[t] = acc >= 0.f ? acc : a2 * acc;
  }
  __syncthreads();
  if (t < 3) {
    float acc = b2[t];
    for (int l = 0; l < NLAT; ++l) acc += sy[l] * w2[l * 3 + t];
    out[q * 3 + t] = acc;
  }
}

// ---------------------------------------------------------------------------
extern "C" void kernel_launch(void* const* d_in, const int* in_sizes, int n_in,
                              void* d_out, int out_size, void* d_ws, size_t ws_size,
                              hipStream_t stream) {
  (void)n_in; (void)out_size; (void)ws_size;
  const float* x    = (const float*)d_in[0];
  const float* mask = (const float*)d_in[1];
  const float* xq   = (const float*)d_in[2];
  const int* edges0 = (const int*)d_in[3];   // A_edges      [2, E]
  const int* edges1 = (const int*)d_in[4];   // A_edges_c_1  [2, E]
  const float* pos  = (const float*)d_in[5]; // merged_nodes [N, 3]
  const int N  = in_sizes[0] / 9;
  const int E  = in_sizes[3] / 2;
  const int NQ = in_sizes[2] / 3;

  // ---- params: jax pytree leaves, dict keys sorted alphabetically ----
  const int P = 9;
  auto agg = [&](int li, int j) { return (const float*)d_in[P + 12 * li + j]; };
  //   agg keys: 0=a1 1=a2 2=a3 3=a4 4=b1 5=b2 6=be 7=bg 8=w1 9=w2 10=we 11=wg
  const int PATT = P + 120;
  auto att = [&](int j) { return (const float*)d_in[PATT + j]; };
  //   att keys: 0=a1 1=a2 2=a3 3=b1 4=b2 5=bc 6=be 7=bv 8=pv 9=w1 10=w2 11=wc 12=we 13=wv
  const int PEMB = PATT + 14;
  const int PEMM = PEMB + 6;
  auto emb = [&](int j) { return (const float*)d_in[PEMB + j]; };  // a1,a2,b1,b2,w1,w2
  auto emm = [&](int j) { return (const float*)d_in[PEMM + j]; };

  // ---- workspace carve-up (~12 MB) ----
  char* wp = (char*)d_ws;
  auto carve = [&](size_t bytes) -> void* {
    void* p = (void*)wp;
    wp += (bytes + 255) & ~(size_t)255;
    return p;
  };
  float* h0   = (float*)carve((size_t)N * 30 * 4);
  float* m0   = (float*)carve((size_t)N * 30 * 4);
  float* bufA = (float*)carve((size_t)N * NHCH * 4);
  float* bufB = (float*)carve((size_t)N * NHCH * 4);
  float* bufS = (float*)carve((size_t)N * NHCH * 4);
  float* sacc = (float*)carve((size_t)N * 15 * 4);
  float* cnt0 = (float*)carve((size_t)N * 4);
  float* cnt1 = (float*)carve((size_t)N * 4);
  float* gp   = (float*)carve(64);
  _Float16* Bt16 = (_Float16*)carve(16 * 96 * 2);
  int* knnIdx = (int*)carve((size_t)NQ * KQ * 4);

  const int nb = (N + 255) / 256;
  const int eb = (E + 255) / 256;
  const int ntiles = (E + 15) / 16;
  const float invN = 1.0f / (float)N;

  embed_kernel<<<nb, 256, 0, stream>>>(x, N, 9, emb(4), emb(2), emb(0),
                                       emb(5), emb(3), emb(1), h0);
  embed_kernel<<<nb, 256, 0, stream>>>(mask, N, 6, emm(4), emm(2), emm(0),
                                       emm(5), emm(3), emm(1), m0);

  hipMemsetAsync(cnt0, 0, (size_t)N * 4, stream);
  hipMemsetAsync(cnt1, 0, (size_t)N * 4, stream);
  count_kernel<<<eb, 256, 0, stream>>>(edges0 + E, E, cnt0);
  count_kernel<<<eb, 256, 0, stream>>>(edges1 + E, E, cnt1);

  auto run_layer = [&](int li, const float* hin, int din, const int* edges,
                       const float* cnt, const float* resid, float* hout) {
    hipMemsetAsync(gp, 0, 16, stream);
    hipMemsetAsync(sacc, 0, (size_t)N * 15 * 4, stream);
    pool_kernel<<<nb, 256, 0, stream>>>(hin, N, din, agg(li, 11), agg(li, 7),
                                        agg(li, 2), gp);
    prep_w1_kernel<<<6, 256, 0, stream>>>(agg(li, 8), din + 48, Bt16);
    if (din == 30) {
      edge_msg_kernel<30><<<512, 256, 0, stream>>>(
          hin, m0, pos, edges, E, ntiles, gp, invN,
          agg(li, 10), agg(li, 6), agg(li, 3), Bt16, agg(li, 4), agg(li, 0), sacc);
    } else {
      edge_msg_kernel<20><<<512, 256, 0, stream>>>(
          hin, m0, pos, edges, E, ntiles, gp, invN,
          agg(li, 10), agg(li, 6), agg(li, 3), Bt16, agg(li, 4), agg(li, 0), sacc);
    }
    node_update_kernel<<<nb, 256, 0, stream>>>(hin, m0, sacc, cnt, gp, invN, N, din,
                                               agg(li, 9), agg(li, 5), agg(li, 1),
                                               resid, hout);
  };

  run_layer(0, h0,   30, edges0, cnt0, nullptr, bufS);  // o1
  run_layer(1, bufS, 20, edges0, cnt0, nullptr, bufA);
  run_layer(2, bufA, 20, edges1, cnt1, bufS,    bufB);  // + o1
  run_layer(3, bufB, 20, edges0, cnt0, nullptr, bufS);  // o2
  run_layer(4, bufS, 20, edges0, cnt0, nullptr, bufA);
  run_layer(5, bufA, 20, edges1, cnt1, bufS,    bufB);  // + o2
  run_layer(6, bufB, 20, edges0, cnt0, nullptr, bufS);  // o3
  run_layer(7, bufS, 20, edges0, cnt0, nullptr, bufA);
  run_layer(8, bufA, 20, edges1, cnt1, bufS,    bufB);  // + o3
  run_layer(9, bufB, 20, edges0, cnt0, nullptr, bufA);  // final node features

  knn_kernel<<<NQ, 32, 0, stream>>>(xq, pos, NQ, N, knnIdx);
  attention_kernel<<<NQ, 128, 0, stream>>>(
      bufA, xq, pos, knnIdx,
      att(8),  /*pv*/  att(12), /*we*/ att(6),  /*be*/
      att(11), /*wc*/  att(5),  /*bc*/ att(13), /*wv*/ att(7), /*bv*/
      att(9),  /*w1*/  att(3),  /*b1*/ att(10), /*w2*/ att(4), /*b2*/
      att(0), att(1), att(2),
      NQ, (float*)d_out);
}
